// STGNN_52682068853355
// MI455X (gfx1250) — compile-verified, matrix operand order checked by reading, not run on hardware
//
#include <hip/hip_runtime.h>
#include <hip/hip_bf16.h>
#include <math.h>

// ---------------- problem constants ----------------
#define NN   1024      // nodes
#define TT   32        // time steps
#define FF   128       // feature dim
#define HH   4         // heads
#define DD   32        // head dim
#define DEG  32        // neighbors per row (fixed)
#define HID  512
#define OUTD 64
#define NT   (NN*TT)   // 32768 tokens
#define TF   (TT*FF)   // 4096

typedef __attribute__((ext_vector_type(16))) _Float16     v16h;
typedef __attribute__((ext_vector_type(8)))  float        v8f;
typedef __attribute__((ext_vector_type(4)))  unsigned int u32x4;
typedef __attribute__((ext_vector_type(8)))  unsigned int u32x8;

// K index inside a 16x32 f16 A/B fragment, per CDNA5 ISA 7.12.2:
// lanes 0-15: VGPR0-3 hold K=0..7, VGPR4-7 hold K=16..23
// lanes 16-31: VGPR0-3 hold K=8..15, VGPR4-7 hold K=24..31
__device__ __forceinline__ int frag_k(int lane, int i) {
  int j  = i >> 1;                    // VGPR index 0..7
  int kb = (lane >> 4) << 3;          // 0 or 8
  int b  = (j < 4) ? (kb + (j << 1)) : (16 + kb + ((j - 4) << 1));
  return b + (i & 1);
}

__device__ __forceinline__ v8f wmma16(v16h a, v16h b, v8f c) {
  return __builtin_amdgcn_wmma_f32_16x16x32_f16(false, a, false, b,
                                                (short)0, c, false, false);
}

// --------- TDM: async 2-D fp32 tile load (global -> LDS), one wave issue -----
// D# per CDNA5 ISA 8.3/8.4: group0 = {count|flags, lds_addr, global_addr, type},
// group1 = {data_size, tensor dims (huge), tile dims, dim0 stride}.
__device__ __forceinline__ void tdm_load_2d(unsigned int lds_addr,
                                            const void* gaddr,
                                            unsigned int tile_d0,
                                            unsigned int tile_d1,
                                            unsigned int stride_elems) {
  unsigned long long ga = (unsigned long long)gaddr;
  u32x4 g0;
  g0[0] = 1u;                                               // count=1, user D#
  g0[1] = lds_addr;                                         // LDS byte address
  g0[2] = (unsigned int)ga;                                 // global_addr[31:0]
  g0[3] = ((unsigned int)(ga >> 32) & 0x01FFFFFFu) | (2u << 30); // [56:32]|type=2
  u32x8 g1;
  g1[0] = 2u << 16;                  // workgroup_mask=0, data_size=2 (4 bytes)
  g1[1] = 0u;                        // no atomic barrier; tensor_dim0[15:0]=0
  g1[2] = 0x4000u;                   // tensor_dim0=2^30 (never OOB)
  g1[3] = 0x4000u | (tile_d0 << 16); // tensor_dim1=2^30 ; tile_dim0
  g1[4] = tile_d1;                   // tile_dim1 ; tile_dim2=0
  g1[5] = stride_elems;              // tensor_dim0_stride[31:0] (element units)
  g1[6] = 0u;                        // stride hi / dim1_stride lo
  g1[7] = 0u;
  asm volatile("tensor_load_to_lds %0, %1" :: "s"(g0), "s"(g1) : "memory");
}

__device__ __forceinline__ unsigned int lds_addr_of(const void* p) {
  return (unsigned int)(unsigned long long)p;   // LDS offset lives in addr[31:0]
}

// ---------------- SpMM: Y[n,:] = sum_d vals[n*DEG+d] * X[cols[n*DEG+d],:] ----
__global__ void spmm_kernel(const int* __restrict__ cols,
                            const float* __restrict__ vals,
                            const float* __restrict__ X,
                            float* __restrict__ Y, int ncols) {
  __shared__ int   sc[DEG];
  __shared__ float sv[DEG];
  int n = blockIdx.x;
  if (threadIdx.x < DEG) {
    sc[threadIdx.x] = cols[n * DEG + threadIdx.x];
    sv[threadIdx.x] = vals[n * DEG + threadIdx.x];
  }
  __syncthreads();
  const float4* X4 = (const float4*)X;
  float4* Y4 = (float4*)Y;
  int nc4 = ncols >> 2;
  for (int c = threadIdx.x; c < nc4; c += blockDim.x) {
    float4 acc = {0.f, 0.f, 0.f, 0.f};
    #pragma unroll 8
    for (int d = 0; d < DEG; ++d) {
      float4 xv = X4[(size_t)sc[d] * nc4 + c];
      float  w  = sv[d];
      acc.x += w * xv.x;  acc.y += w * xv.y;
      acc.z += w * xv.z;  acc.w += w * xv.w;
    }
    Y4[(size_t)n * nc4 + c] = acc;
  }
}

// ---------------- generic WMMA GEMM: C = act(A[M,K] x W(^T) + bias) ----------
// wTrans=1: W is [Nout,K] (PyTorch x@W.T);  wTrans=0: W is [K,Nout] (x@W)
// 256 threads = 8 waves; each wave -> one 16x16 C tile.
// A tile (16x32, shared by block) and per-wave W tile arrive via TDM async DMA.
__global__ void wmma_gemm(const float* __restrict__ A, const float* __restrict__ W,
                          const float* __restrict__ bias, float* __restrict__ C,
                          int M, int K, int Nout, int lda, int ldc,
                          int wTrans, int act) {
  __shared__ float sA[16 * 32];        // block-shared A tile
  __shared__ float sW[8][16 * 32];     // per-wave W tile
  int tid  = threadIdx.x;
  int lane = tid & 31;
  int wid  = tid >> 5;
  int rowbase = blockIdx.y * 16;
  int colbase = (blockIdx.x * 8 + wid) * 16;
  bool active = colbase < Nout;
  int r = lane & 15;
  unsigned int ldsA = lds_addr_of(sA);
  unsigned int ldsW = lds_addr_of(&sW[wid][0]);
  v8f acc = {};
  for (int kb = 0; kb < K; kb += 32) {
    __syncthreads();                                     // prior sA reads done
    if (wid == 0)                                        // one TDM for A tile
      tdm_load_2d(ldsA, A + (size_t)rowbase * lda + kb, 32u, 16u, (unsigned)lda);
    if (active) {                                        // per-wave W tile
      if (wTrans)   // rows = 16 output cols, contiguous K: tile 32x16, stride K
        tdm_load_2d(ldsW, W + (size_t)colbase * K + kb, 32u, 16u, (unsigned)K);
      else          // rows = 32 K values, contiguous N: tile 16x32, stride Nout
        tdm_load_2d(ldsW, W + (size_t)kb * Nout + colbase, 16u, 32u, (unsigned)Nout);
    }
    __builtin_amdgcn_s_wait_tensorcnt(0);                // our DMAs landed
    __syncthreads();                                     // sA visible block-wide
    if (active) {
      v16h a, b;
      #pragma unroll
      for (int i = 0; i < 16; ++i) {
        int k = frag_k(lane, i);
        a[i] = (_Float16)sA[r * 32 + k];
        b[i] = (_Float16)(wTrans ? sW[wid][r * 32 + k]   // [n][k]
                                 : sW[wid][k * 16 + r]); // [k][n]
      }
      acc = wmma16(a, b, acc);
    }
  }
  if (active) {
    int col = colbase + r;
    float bv = bias ? bias[col] : 0.f;
    #pragma unroll
    for (int i = 0; i < 8; ++i) {
      int row = rowbase + i + ((lane >> 4) << 3);
      float v = acc[i] + bv;
      if (act == 1) v = fmaxf(v, 0.f);
      C[(size_t)row * ldc + col] = v;
    }
  }
}

// ---------------- GRU gate fusion -------------------------------------------
__global__ void gru_gate(const float* __restrict__ gi, const float* __restrict__ gh,
                         const float* __restrict__ h, float* __restrict__ g_out,
                         float* __restrict__ hm_slot) {
  int idx = blockIdx.x * blockDim.x + threadIdx.x;
  if (idx >= NN * FF) return;
  int n = idx >> 7, f = idx & (FF - 1);
  const float* gin = gi + (size_t)n * (3 * FF);
  const float* ghn = gh + (size_t)n * (3 * FF);
  float rg = 1.f / (1.f + __expf(-(gin[f]        + ghn[f])));
  float zg = 1.f / (1.f + __expf(-(gin[FF + f]   + ghn[FF + f])));
  float ng = tanhf(gin[2 * FF + f] + rg * ghn[2 * FF + f]);
  float val = (1.f - zg) * ng + zg * h[idx];
  g_out[idx] = val;
  hm_slot[(size_t)n * TF + f] = val;     // Hm[:, t, :]
}

// ---------------- positional encoding add -----------------------------------
__global__ void pos_add(float* __restrict__ Hm) {
  int idx = blockIdx.x * blockDim.x + threadIdx.x;   // over N*T
  if (idx >= NN * TT) return;
  int n = idx / TT, t = idx % TT;
  float inv = __expf(-(2.0f * (float)n / (float)FF) * 9.210340371976184f);
  float ang = (float)t * inv;
  float pe  = ((t & 1) == 0) ? __sinf(ang) : __cosf(ang);
  float* p = Hm + (size_t)idx * FF;
  for (int f = 0; f < FF; ++f) p[f] += pe;
}

// ---------------- fused flash attention (per wave: 16 queries of one (t,h)) --
// qkv: [NT, 3F] rows in (n,t) order; ctx: [NT, F]
__global__ void flash_attn(const float* __restrict__ qkv, float* __restrict__ ctx) {
  __shared__ _Float16 sP[8][16 * 32];     // wave-private P staging
  int tid = threadIdx.x, lane = tid & 31, wid = tid >> 5;
  int t = blockIdx.y, h = blockIdx.z;
  int qbase = (blockIdx.x * 8 + wid) * 16;
  const int ld = 3 * FF;
  const float scale = 0.17677669529663687f;   // 1/sqrt(32)
  int r = lane & 15;
  _Float16* myP = &sP[wid][0];

  v16h qa;
  #pragma unroll
  for (int i = 0; i < 16; ++i) {
    int k = frag_k(lane, i);
    qa[i] = (_Float16)(qkv[(size_t)((qbase + r) * TT + t) * ld + h * DD + k] * scale);
  }
  float mI[8], lI[8];
  v8f c0 = {}, c1 = {};
  #pragma unroll
  for (int i = 0; i < 8; ++i) { mI[i] = -1e30f; lI[i] = 0.f; }

  for (int m0 = 0; m0 < NN; m0 += 32) {
    // S = Q @ K^T for keys m0..m0+31 (two 16-key column tiles)
    v16h kb0, kb1;
    #pragma unroll
    for (int i = 0; i < 16; ++i) {
      int k = frag_k(lane, i);
      kb0[i] = (_Float16)qkv[(size_t)((m0 + r)      * TT + t) * ld + FF + h * DD + k];
      kb1[i] = (_Float16)qkv[(size_t)((m0 + 16 + r) * TT + t) * ld + FF + h * DD + k];
    }
    v8f z = {};
    v8f s0 = wmma16(qa, kb0, z);
    v8f s1 = wmma16(qa, kb1, z);

    // online softmax; each half-wave owns its 8 rows (C layout: M = i + 8*(lane>=16))
    #pragma unroll
    for (int i = 0; i < 8; ++i) {
      float tm = fmaxf(s0[i], s1[i]);
      for (int off = 8; off >= 1; off >>= 1) tm = fmaxf(tm, __shfl_xor(tm, off, 16));
      float nm = fmaxf(mI[i], tm);
      float sc = __expf(mI[i] - nm);
      float p0 = __expf(s0[i] - nm);
      float p1 = __expf(s1[i] - nm);
      float rs = p0 + p1;
      for (int off = 8; off >= 1; off >>= 1) rs += __shfl_xor(rs, off, 16);
      lI[i] = lI[i] * sc + rs;
      mI[i] = nm;
      c0[i] *= sc;  c1[i] *= sc;
      int prow = i + ((lane >> 4) << 3);
      myP[prow * 32 +       (lane & 15)] = (_Float16)p0;
      myP[prow * 32 + 16 +  (lane & 15)] = (_Float16)p1;
    }
    // ctx += P(16x32) @ V(32x32): re-shape P via LDS into an A fragment
    v16h pa, v0, v1;
    #pragma unroll
    for (int i = 0; i < 16; ++i) {
      int k = frag_k(lane, i);                 // k = key index within tile
      pa[i] = myP[r * 32 + k];
      v0[i] = (_Float16)qkv[(size_t)((m0 + k) * TT + t) * ld + 2 * FF + h * DD + r];
      v1[i] = (_Float16)qkv[(size_t)((m0 + k) * TT + t) * ld + 2 * FF + h * DD + 16 + r];
    }
    c0 = wmma16(pa, v0, c0);
    c1 = wmma16(pa, v1, c1);
  }
  #pragma unroll
  for (int i = 0; i < 8; ++i) {
    int row = qbase + i + ((lane >> 4) << 3);
    float inv = 1.f / lI[i];
    ctx[(size_t)(row * TT + t) * FF + h * DD +       (lane & 15)] = c0[i] * inv;
    ctx[(size_t)(row * TT + t) * FF + h * DD + 16 +  (lane & 15)] = c1[i] * inv;
  }
}

// ---------------- residual + layernorm (cols == blockDim == 128) -------------
__global__ void residual_ln(const float* __restrict__ a, const float* __restrict__ b,
                            const float* __restrict__ g, const float* __restrict__ be,
                            float* __restrict__ out) {
  __shared__ float red[4];
  int row = blockIdx.x, tid = threadIdx.x;
  float x = a[(size_t)row * FF + tid] + (b ? b[(size_t)row * FF + tid] : 0.f);
  float s = x;
  for (int off = 16; off >= 1; off >>= 1) s += __shfl_xor(s, off, 32);
  if ((tid & 31) == 0) red[tid >> 5] = s;
  __syncthreads();
  float mu = (red[0] + red[1] + red[2] + red[3]) * (1.f / FF);
  float d = x - mu;
  float vs = d * d;
  for (int off = 16; off >= 1; off >>= 1) vs += __shfl_xor(vs, off, 32);
  __syncthreads();
  if ((tid & 31) == 0) red[tid >> 5] = vs;
  __syncthreads();
  float var = (red[0] + red[1] + red[2] + red[3]) * (1.f / FF);
  out[(size_t)row * FF + tid] = d * rsqrtf(var + 1e-5f) * g[tid] + be[tid];
}

__global__ void fill0(float* p, int n) {
  int i = blockIdx.x * blockDim.x + threadIdx.x;
  if (i < n) p[i] = 0.f;
}

// ---------------- host orchestration ----------------------------------------
extern "C" void kernel_launch(void* const* d_in, const int* in_sizes, int n_in,
                              void* d_out, int out_size, void* d_ws, size_t ws_size,
                              hipStream_t stream) {
  (void)in_sizes; (void)n_in; (void)out_size; (void)ws_size;
  const int*   A_cols    = (const int*)  d_in[1];
  const float* A_vals    = (const float*)d_in[2];
  const float* X         = (const float*)d_in[3];
  const float* W_gcn     = (const float*)d_in[4];
  const float* W_ih      = (const float*)d_in[5];
  const float* W_hh      = (const float*)d_in[6];
  const float* b_ih      = (const float*)d_in[7];
  const float* b_hh      = (const float*)d_in[8];
  const float* in_proj_w = (const float*)d_in[9];
  const float* in_proj_b = (const float*)d_in[10];
  const float* out_proj_w= (const float*)d_in[11];
  const float* out_proj_b= (const float*)d_in[12];
  const float* ln1_g = (const float*)d_in[13];
  const float* ln1_b = (const float*)d_in[14];
  const float* ln2_g = (const float*)d_in[15];
  const float* ln2_b = (const float*)d_in[16];
  const float* lnf_g = (const float*)d_in[17];
  const float* lnf_b = (const float*)d_in[18];
  const float* ff1_w = (const float*)d_in[19];
  const float* ff1_b = (const float*)d_in[20];
  const float* ff2_w = (const float*)d_in[21];
  const float* ff2_b = (const float*)d_in[22];
  const float* fc1_w = (const float*)d_in[23];
  const float* fc1_b = (const float*)d_in[24];
  const float* fc2_w = (const float*)d_in[25];
  const float* fc2_b = (const float*)d_in[26];
  const float* fc3_w = (const float*)d_in[27];
  const float* fc3_b = (const float*)d_in[28];
  float* out = (float*)d_out;

  // workspace layout (aliased; peak ~171 MB)
  float* ws = (float*)d_ws;
  size_t o = 0;
  auto alloc = [&](size_t n){ float* p = ws + o; o += n; return p; };
  float* bufA = alloc((size_t)NN * TF);       // spmm tmp -> ctx -> ff_out
  float* bufB = alloc((size_t)NN * TF);       // X_tilde -> attn_out
  float* bufC = alloc((size_t)NN * TF);       // Hm -> enc
  float* bufD = alloc((size_t)3 * NN * TF);   // qkv -> {x1, x2}
  float* bufE = alloc((size_t)NT * HID);      // ffh -> {h1, h2}
  float* hb   = alloc((size_t)NN * FF);
  float* gb   = alloc((size_t)NN * FF);
  float* gi   = alloc((size_t)NN * 3 * FF);
  float* gh   = alloc((size_t)NN * 3 * FF);
  float* sp   = alloc((size_t)NN * FF);

  auto gemm = [&](const float* A, int lda, const float* W, int wT, const float* bias,
                  float* C, int ldc, int M, int K, int Nout, int act) {
    dim3 g((Nout + 127) / 128, M / 16);
    wmma_gemm<<<g, dim3(256), 0, stream>>>(A, W, bias, C, M, K, Nout, lda, ldc, wT, act);
  };

  // 1) X_tilde = relu(spmm(X) @ W_gcn)
  spmm_kernel<<<NN, 256, 0, stream>>>(A_cols, A_vals, X, bufA, TF);
  gemm(bufA, FF, W_gcn, 0, nullptr, bufB, FF, NT, FF, FF, 1);

  // 2) sequential GRU + per-step GCN; Hs goes into Hm[:, t, :]
  fill0<<<(NN * FF + 255) / 256, 256, 0, stream>>>(hb, NN * FF);
  for (int t = 0; t < TT; ++t) {
    gemm(bufB + (size_t)t * FF, TF, W_ih, 1, b_ih, gi, 3 * FF, NN, FF, 3 * FF, 0);
    gemm(hb, FF, W_hh, 1, b_hh, gh, 3 * FF, NN, FF, 3 * FF, 0);
    gru_gate<<<(NN * FF + 255) / 256, 256, 0, stream>>>(gi, gh, hb, gb,
                                                        bufC + (size_t)t * FF);
    spmm_kernel<<<NN, 256, 0, stream>>>(A_cols, A_vals, gb, sp, FF);
    gemm(sp, FF, W_gcn, 0, nullptr, hb, FF, NN, FF, FF, 1);
  }
  pos_add<<<(NN * TT + 255) / 256, 256, 0, stream>>>(bufC);

  // 3) transformer encoder layer
  gemm(bufC, FF, in_proj_w, 1, in_proj_b, bufD, 3 * FF, NT, FF, 3 * FF, 0);   // qkv
  flash_attn<<<dim3(NN / 16 / 8, TT, HH), 256, 0, stream>>>(bufD, bufA);      // ctx
  gemm(bufA, FF, out_proj_w, 1, out_proj_b, bufB, FF, NT, FF, FF, 0);         // attn_out
  float* x1 = bufD;
  float* x2 = bufD + (size_t)NN * TF;
  residual_ln<<<NT, FF, 0, stream>>>(bufC, bufB, ln1_g, ln1_b, x1);
  gemm(x1, FF, ff1_w, 1, ff1_b, bufE, HID, NT, FF, HID, 1);                   // ffh
  gemm(bufE, HID, ff2_w, 1, ff2_b, bufA, FF, NT, HID, FF, 0);                 // ff_out
  residual_ln<<<NT, FF, 0, stream>>>(x1, bufA, ln2_g, ln2_b, x2);
  residual_ln<<<NT, FF, 0, stream>>>(x2, nullptr, lnf_g, lnf_b, bufC);        // enc

  // 4) MLP head: enc viewed as [N, T*F]
  float* h1 = bufE;
  float* h2 = bufE + (size_t)NN * HID;
  gemm(bufC, TF, fc1_w, 1, fc1_b, h1, HID, NN, TF, HID, 1);
  gemm(h1, HID, fc2_w, 1, fc2_b, h2, HID, NN, HID, HID, 1);
  gemm(h2, HID, fc3_w, 1, fc3_b, out, OUTD, NN, HID, OUTD, 0);
}
